// RefineNet_16587163697667
// MI455X (gfx1250) — compile-verified, matrix operand order checked by reading, not run on hardware
//
#include <hip/hip_runtime.h>
#include <hip/hip_bf16.h>
#include <math.h>

// ---------------- problem constants (from reference) ----------------
#define N_NODES   50000
#define E_RAW     400000
#define E_TOT     (E_RAW + N_NODES)   // self-loops appended
#define PHYS_DIM  128
#define SEM_DIM   17
#define HID       64
#define HEADS     4
#define NCLS      17
#define NEG_SLOPE 0.2f
#define BN_EPS    1e-5f

typedef __attribute__((ext_vector_type(16))) _Float16 v16h;
typedef __attribute__((ext_vector_type(8)))  _Float16 v8h;
typedef __attribute__((ext_vector_type(8)))  float    v8f;

// Order-preserving float<->int involution for integer atomicMax segment-max.
__device__ __forceinline__ int ford(float f) {
  unsigned u = __float_as_uint(f);
  return (int)((u & 0x80000000u) ? (0x80000000u - u) : u);
}
__device__ __forceinline__ float forddec(int k) {
  unsigned u = (unsigned)k;
  return __uint_as_float((u & 0x80000000u) ? (0x80000000u - u) : u);
}

// ---------------- packing kernels (guards live here, not in GEMM) ----------------
// A[M,*] (row stride lda) f32 -> Ah[M,Kp] f16, zero padded (Kp multiple of 32).
__global__ void k_pack_a(const float* __restrict__ A, _Float16* __restrict__ Ah,
                         int M, int K, int Kp, int lda) {
  int t = blockIdx.x * blockDim.x + threadIdx.x;
  if (t >= M * Kp) return;
  const int row = t / Kp, j = t - row * Kp;
  Ah[t] = (j < K) ? (_Float16)A[(size_t)row * lda + j] : (_Float16)0.0f;
}

// B[K,N] f32 -> Bp f16 in WMMA B-fragment order: [kt][nt][lane 0..31][j 0..15].
// lane l covers col = nt*16 + (l&15); half j covers k = kt*32 + 8*(l>>4) + (j<8 ? j : 8+j).
// Out-of-range rows/cols -> 0, so padded N-tiles contribute nothing.
__global__ void k_pack_b(const float* __restrict__ B, _Float16* __restrict__ Bp,
                         int K, int N, int ntiles, int total) {
  int t = blockIdx.x * blockDim.x + threadIdx.x;
  if (t >= total) return;
  const int j    = t & 15;
  const int lane = (t >> 4) & 31;
  const int rest = t >> 9;
  const int nt   = rest % ntiles;
  const int kt   = rest / ntiles;
  const int col  = nt * 16 + (lane & 15);
  const int k    = kt * 32 + ((lane >> 4) << 3) + ((j < 8) ? j : (8 + j));
  Bp[t] = (k < K && col < N) ? (_Float16)B[(size_t)k * N + col] : (_Float16)0.0f;
}

// ------------------------------------------------------------------
// Branch-free WMMA GEMM, 16x64 output per wave: 4 N-tiles share one
// A fragment (4x less A traffic) and run as 4 independent WMMAs per
// K-step.  Np is always a multiple of 64 -> no guards anywhere, so
// loads clause together and waits batch.  Group id is wave-uniform
// -> EXEC all-ones for every WMMA (ISA 7.12 requirement).
// ------------------------------------------------------------------
__global__ void gemm_wmma(const _Float16* __restrict__ Ah, const _Float16* __restrict__ Bp,
                          float* __restrict__ C, int mtiles, int ngroups, int ktiles,
                          int Kp, int Np) {
  const int wave = threadIdx.x >> 5;
  const int lane = threadIdx.x & 31;
  const int gid  = blockIdx.x * 4 + wave;
  if (gid >= mtiles * ngroups) return;            // wave-uniform exit
  const int mt   = gid / ngroups;
  const int nt0  = (gid - mt * ngroups) << 2;
  const int rc   = lane & 15;
  const int koff = (lane >> 4) << 3;              // 0 or 8
  const int ntiles = ngroups << 2;

  const _Float16* arow  = Ah + (size_t)(mt * 16 + rc) * Kp + koff;
  const v16h*     bbase = (const v16h*)Bp + (size_t)nt0 * 32 + lane;
  const size_t    bstep = (size_t)ntiles * 32;    // v16h units per kt

  v8f acc0 = {}, acc1 = {}, acc2 = {}, acc3 = {};
  for (int kt = 0; kt < ktiles; ++kt) {
    if (kt + 1 < ktiles) __builtin_prefetch(arow + (size_t)(kt + 1) * 32, 0, 3);
    const v8h alo = *(const v8h*)(arow + (size_t)kt * 32);
    const v8h ahi = *(const v8h*)(arow + (size_t)kt * 32 + 16);
    const v16h af = __builtin_shufflevector(alo, ahi,
        0, 1, 2, 3, 4, 5, 6, 7, 8, 9, 10, 11, 12, 13, 14, 15);
    const v16h* bk = bbase + (size_t)kt * bstep;
    const v16h b0 = bk[0];
    const v16h b1 = bk[32];
    const v16h b2 = bk[64];
    const v16h b3 = bk[96];
    acc0 = __builtin_amdgcn_wmma_f32_16x16x32_f16(false, af, false, b0, (short)0, acc0, false, false);
    acc1 = __builtin_amdgcn_wmma_f32_16x16x32_f16(false, af, false, b1, (short)0, acc1, false, false);
    acc2 = __builtin_amdgcn_wmma_f32_16x16x32_f16(false, af, false, b2, (short)0, acc2, false, false);
    acc3 = __builtin_amdgcn_wmma_f32_16x16x32_f16(false, af, false, b3, (short)0, acc3, false, false);
  }
  float* crow = C + (size_t)(mt * 16 + koff) * Np + nt0 * 16 + rc;
#pragma unroll
  for (int v = 0; v < 8; ++v) crow[(size_t)v * Np +  0] = acc0[v];
#pragma unroll
  for (int v = 0; v < 8; ++v) crow[(size_t)v * Np + 16] = acc1[v];
#pragma unroll
  for (int v = 0; v < 8; ++v) crow[(size_t)v * Np + 32] = acc2[v];
#pragma unroll
  for (int v = 0; v < 8; ++v) crow[(size_t)v * Np + 48] = acc3[v];
}

// ---------------- graph / elementwise kernels ----------------
__global__ void k_build_edges(const int* __restrict__ ei,
                              int* __restrict__ srcs, int* __restrict__ dsts) {
  int t = blockIdx.x * blockDim.x + threadIdx.x;
  if (t >= E_TOT) return;
  if (t < E_RAW) { srcs[t] = ei[t]; dsts[t] = ei[E_RAW + t]; }
  else           { srcs[t] = t - E_RAW; dsts[t] = t - E_RAW; }
}

__global__ void k_zero(float* __restrict__ p, int n) {
  int t = blockIdx.x * blockDim.x + threadIdx.x;
  if (t < n) p[t] = 0.0f;
}

__global__ void k_alpha_init(const float* __restrict__ h,
                             const float* __restrict__ a_src,
                             const float* __restrict__ a_dst,
                             float* __restrict__ as_, float* __restrict__ ad_,
                             int* __restrict__ m_int, float* __restrict__ denom,
                             int H, int C) {
  int t = blockIdx.x * blockDim.x + threadIdx.x;
  if (t >= N_NODES * H) return;
  const int n = t / H, hh = t - n * H;
  const float4* hp = (const float4*)(h + (size_t)n * H * C + (size_t)hh * C);
  const float4* s  = (const float4*)(a_src + (size_t)hh * C);
  const float4* d  = (const float4*)(a_dst + (size_t)hh * C);
  float va = 0.0f, vd = 0.0f;
  for (int c = 0; c < C / 4; ++c) {
    const float4 hv = hp[c], sv = s[c], dv = d[c];
    va += hv.x * sv.x + hv.y * sv.y + hv.z * sv.z + hv.w * sv.w;
    vd += hv.x * dv.x + hv.y * dv.y + hv.z * dv.z + hv.w * dv.w;
  }
  as_[t]   = va;
  ad_[t]   = vd;
  m_int[t] = ford(-3.0e38f);
  denom[t] = 0.0f;
}

__global__ void k_edge_logits(const int* __restrict__ srcs, const int* __restrict__ dsts,
                              const float* __restrict__ as_, const float* __restrict__ ad_,
                              float* __restrict__ evals, int* __restrict__ m_int, int H) {
  int t = blockIdx.x * blockDim.x + threadIdx.x;
  if (t >= E_TOT * H) return;
  const int e = t / H, hh = t - e * H;
  float v = as_[srcs[e] * H + hh] + ad_[dsts[e] * H + hh];
  v = (v > 0.0f) ? v : NEG_SLOPE * v;            // leaky_relu
  evals[t] = v;
  atomicMax(&m_int[dsts[e] * H + hh], ford(v));
}

__global__ void k_edge_exp(const int* __restrict__ dsts, float* __restrict__ evals,
                           const int* __restrict__ m_int, float* __restrict__ denom, int H) {
  int t = blockIdx.x * blockDim.x + threadIdx.x;
  if (t >= E_TOT * H) return;
  const int e = t / H, hh = t - e * H;
  const float m  = forddec(m_int[dsts[e] * H + hh]);   // finite (self-loops)
  const float ex = expf(evals[t] - m);
  evals[t] = ex;
  atomicAdd(&denom[dsts[e] * H + hh], ex);
}

// normalize once per (edge,head) so the big scatter has no division
__global__ void k_edge_alpha(const int* __restrict__ dsts, float* __restrict__ evals,
                             const float* __restrict__ denom, int H) {
  int t = blockIdx.x * blockDim.x + threadIdx.x;
  if (t >= E_TOT * H) return;
  const int e = t / H, hh = t - e * H;
  evals[t] = evals[t] / (denom[dsts[e] * H + hh] + 1e-16f);
}

// one thread per (edge, 4 channels): b128 gather + 4 f32 atomics
__global__ void k_edge_agg(const int* __restrict__ srcs, const int* __restrict__ dsts,
                           const float* __restrict__ h, const float* __restrict__ alpha,
                           float* __restrict__ out, int H, int C, int total4) {
  int t = blockIdx.x * blockDim.x + threadIdx.x;
  if (t >= total4) return;
  const int D  = H * C;
  const int D4 = D >> 2;
  const int e  = t / D4;
  const int r  = (t - e * D4) << 2;              // channel quad start
  const int hh = r / C;
  const float a  = alpha[e * H + hh];
  const float4 hv = *(const float4*)(h + (size_t)srcs[e] * D + r);
  float* o = out + (size_t)dsts[e] * D + r;
  atomicAdd(o + 0, hv.x * a);
  atomicAdd(o + 1, hv.y * a);
  atomicAdd(o + 2, hv.z * a);
  atomicAdd(o + 3, hv.w * a);
}

__global__ void k_bias_bn_elu(const float* __restrict__ in, const float* __restrict__ b,
                              const float* __restrict__ g, const float* __restrict__ be,
                              const float* __restrict__ mu, const float* __restrict__ va,
                              float* __restrict__ out, int D) {
  int t = blockIdx.x * blockDim.x + threadIdx.x;
  if (t >= N_NODES * D) return;
  const int j = t % D;
  float v = in[t] + b[j];
  v = (v - mu[j]) * (g[j] * rsqrtf(va[j] + BN_EPS)) + be[j];
  out[t] = (v > 0.0f) ? v : expm1f(v);
}

__global__ void k_gate_fuse(float* __restrict__ hp, const float* __restrict__ hs,
                            const float* __restrict__ Wg, const float* __restrict__ bg) {
  int n = blockIdx.x * blockDim.x + threadIdx.x;
  if (n >= N_NODES) return;
  float acc = bg[0];
  const size_t base = (size_t)n * HID;
  for (int j = 0; j < HID; ++j) acc += hp[base + j] * Wg[j];
  for (int j = 0; j < HID; ++j) acc += hs[base + j] * Wg[HID + j];
  const float z = 1.0f / (1.0f + expf(-acc));
  for (int j = 0; j < HID; ++j)
    hp[base + j] = z * hp[base + j] + (1.0f - z) * hs[base + j];
}

// hd stored with row stride `stride`, D valid channels
__global__ void k_bias_relu(float* __restrict__ x, const float* __restrict__ b,
                            int D, int stride) {
  int t = blockIdx.x * blockDim.x + threadIdx.x;
  if (t >= N_NODES * D) return;
  const int n = t / D, j = t - n * D;
  const size_t i = (size_t)n * stride + j;
  const float v = x[i] + b[j];
  x[i] = (v > 0.0f) ? v : 0.0f;
}

// logits stored with padded row stride -> compact write to d_out
__global__ void k_bias_out(const float* __restrict__ in, const float* __restrict__ b,
                           float* __restrict__ out, int stride) {
  int t = blockIdx.x * blockDim.x + threadIdx.x;
  if (t >= N_NODES * NCLS) return;
  const int n = t / NCLS, j = t - n * NCLS;
  out[t] = in[(size_t)n * stride + j] + b[j];
}

// ---------------- host-side orchestration ----------------
struct Enc {
  const float *W1,*as1,*ad1,*b1,*g1,*be1,*mu1,*va1;
  const float *W2,*as2,*ad2,*b2,*g2,*be2,*mu2,*va2;
};

static inline dim3 grid1(int n, int bs = 256) { return dim3((unsigned)((n + bs - 1) / bs)); }

// C[M,Np] = A[M,K] @ B[K,N] via pack->WMMA.  Np = N rounded up to 64
// (so every wave owns 4 full N-tiles; padded tiles are zeros).
static void gemm(const float* A, const float* B, float* C,
                 int M, int N, int K, int lda,
                 _Float16* ah, _Float16* bp, hipStream_t stream) {
  const int Kp = (K + 31) & ~31;
  const int Np = (N + 63) & ~63;
  const int mtiles = M / 16, ntiles = Np / 16, ktiles = Kp / 32;
  k_pack_a<<<grid1(M * Kp), 256, 0, stream>>>(A, ah, M, K, Kp, lda);
  const int btot = ktiles * ntiles * 512;
  k_pack_b<<<grid1(btot), 256, 0, stream>>>(B, bp, K, N, ntiles, btot);
  const int ngroups = ntiles >> 2;
  const int waves   = mtiles * ngroups;
  gemm_wmma<<<grid1(waves, 4), 128, 0, stream>>>(ah, bp, C, mtiles, ngroups, ktiles, Kp, Np);
}

static void gat_layer(const float* x, int in_dim,
                      const float* W, const float* a_src, const float* a_dst,
                      const float* b, const float* g, const float* be,
                      const float* mu, const float* va, int H, int C,
                      float* h_buf, float* agg_buf, float* out_buf,
                      float* as_, float* ad_, int* m_int, float* denom, float* evals,
                      const int* srcs, const int* dsts,
                      _Float16* ah, _Float16* bp, hipStream_t stream) {
  const int D = H * C;   // 256 or 64: multiples of 64 -> Np == D, h_buf rows tight
  gemm(x, W, h_buf, N_NODES, D, in_dim, in_dim, ah, bp, stream);
  k_alpha_init<<<grid1(N_NODES * H), 256, 0, stream>>>(h_buf, a_src, a_dst, as_, ad_, m_int, denom, H, C);
  k_edge_logits<<<grid1(E_TOT * H), 256, 0, stream>>>(srcs, dsts, as_, ad_, evals, m_int, H);
  k_edge_exp<<<grid1(E_TOT * H), 256, 0, stream>>>(dsts, evals, m_int, denom, H);
  k_edge_alpha<<<grid1(E_TOT * H), 256, 0, stream>>>(dsts, evals, denom, H);
  k_zero<<<grid1(N_NODES * D), 256, 0, stream>>>(agg_buf, N_NODES * D);
  const int total4 = E_TOT * (D >> 2);                 // <= 28.8M threads
  k_edge_agg<<<grid1(total4), 256, 0, stream>>>(srcs, dsts, h_buf, evals, agg_buf, H, C, total4);
  k_bias_bn_elu<<<grid1(N_NODES * D), 256, 0, stream>>>(agg_buf, b, g, be, mu, va, out_buf, D);
}

static void run_encoder(const float* x, int in_dim, const Enc& p, float* dest,
                        float* bufA, float* bufB,
                        float* as_, float* ad_, int* m_int, float* denom, float* evals,
                        const int* srcs, const int* dsts,
                        _Float16* ah, _Float16* bp, hipStream_t stream) {
  // layer 1: H=4, C=64 (concat, D=256): h->bufA, agg->bufB, elu-out->bufA
  gat_layer(x, in_dim, p.W1, p.as1, p.ad1, p.b1, p.g1, p.be1, p.mu1, p.va1,
            HEADS, HID, bufA, bufB, bufA, as_, ad_, m_int, denom, evals, srcs, dsts, ah, bp, stream);
  // layer 2: H=1, C=64 (mean over one head == identity): h->bufB, agg->bufA, out->dest
  gat_layer(bufA, HEADS * HID, p.W2, p.as2, p.ad2, p.b2, p.g2, p.be2, p.mu2, p.va2,
            1, HID, bufB, bufA, dest, as_, ad_, m_int, denom, evals, srcs, dsts, ah, bp, stream);
}

extern "C" void kernel_launch(void* const* d_in, const int* in_sizes, int n_in,
                              void* d_out, int out_size, void* d_ws, size_t ws_size,
                              hipStream_t stream) {
  // ---- inputs (setup_inputs() recursive dict-insertion order) ----
  const float* x_phys     = (const float*)d_in[0];   // [50000,128]
  const float* x_sem      = (const float*)d_in[1];   // [50000,17]
  const int*   edge_index = (const int*)  d_in[2];   // [2,400000]
  auto F = [&](int i) { return (const float*)d_in[i]; };
  auto load_enc = [&](int i0) {
    Enc e;
    e.W1 = F(i0);     e.as1 = F(i0+1);  e.ad1 = F(i0+2);  e.b1 = F(i0+3);
    e.g1 = F(i0+4);   e.be1 = F(i0+5);  e.mu1 = F(i0+6);  e.va1 = F(i0+7);
    e.W2 = F(i0+8);   e.as2 = F(i0+9);  e.ad2 = F(i0+10); e.b2 = F(i0+11);
    e.g2 = F(i0+12);  e.be2 = F(i0+13); e.mu2 = F(i0+14); e.va2 = F(i0+15);
    return e;
  };
  const Enc phys = load_enc(3);
  const Enc sem  = load_enc(19);
  const float* Wg  = F(35); const float* bg  = F(36);
  const float* Wd1 = F(37); const float* bd1 = F(38);
  const float* Wd2 = F(39); const float* bd2 = F(40);

  // ---- workspace carving (all regions rewritten every call; 16B aligned) ----
  int*      srcs  = (int*)d_ws;                              // E_TOT
  int*      dsts  = srcs + E_TOT;                            // E_TOT
  float*    bufA  = (float*)(dsts + E_TOT);                  // 50000*256 f32
  float*    bufB  = bufA + (size_t)N_NODES * 256;            // 50000*256 f32
  _Float16* ah    = (_Float16*)(bufB + (size_t)N_NODES * 256); // 50000*256 f16
  _Float16* bp    = ah + (size_t)N_NODES * 256;              // 65536 f16
  float*    as_   = (float*)(bp + 65536);                    // 50000*4
  float*    ad_   = as_  + (size_t)N_NODES * HEADS;          // 50000*4
  int*      m_int = (int*)(ad_ + (size_t)N_NODES * HEADS);   // 50000*4
  float*    denom = (float*)(m_int + (size_t)N_NODES * HEADS); // 50000*4
  float*    evals = denom + (size_t)N_NODES * HEADS;         // E_TOT*4
  float*    hp    = evals + (size_t)E_TOT * HEADS;           // 50000*64
  float*    hs    = hp + (size_t)N_NODES * HID;              // 50000*64

  // ---- pipeline ----
  k_build_edges<<<grid1(E_TOT), 256, 0, stream>>>(edge_index, srcs, dsts);

  run_encoder(x_phys, PHYS_DIM, phys, hp, bufA, bufB, as_, ad_, m_int, denom, evals,
              srcs, dsts, ah, bp, stream);
  run_encoder(x_sem,  SEM_DIM,  sem,  hs, bufA, bufB, as_, ad_, m_int, denom, evals,
              srcs, dsts, ah, bp, stream);

  // gated fusion (in place into hp)
  k_gate_fuse<<<grid1(N_NODES), 256, 0, stream>>>(hp, hs, Wg, bg);

  // decoder: hd = relu(fused @ Wd1 + bd1) -> bufB, row stride 64 (Np), 32 valid
  gemm(hp, Wd1, bufB, N_NODES, 32, HID, HID, ah, bp, stream);
  k_bias_relu<<<grid1(N_NODES * 32), 256, 0, stream>>>(bufB, bd1, 32, 64);

  // logits = hd @ Wd2 + bd2 : A has stride 64 / K=32; C -> bufA stride 64
  gemm(bufB, Wd2, bufA, N_NODES, NCLS, 32, 64, ah, bp, stream);
  k_bias_out<<<grid1(N_NODES * NCLS), 256, 0, stream>>>(bufA, bd2, (float*)d_out, 64);
}
